// attention_mech_deep_52888227283043
// MI455X (gfx1250) — matrix operand order repaired and checked by reading, not compiled, not run on hardware
//
#include <hip/hip_runtime.h>
#include <hip/hip_bf16.h>

typedef __attribute__((ext_vector_type(16))) _Float16 v16h;
typedef __attribute__((ext_vector_type(8)))  float    v8f;

#define DEVI __device__ __forceinline__

constexpr int BC  = 64;   // B*C
constexpr int Nn  = 64;   // N == M
constexpr int Ss  = 128;  // S
constexpr int IRc = 96;   // IR == IC
constexpr float EPSf = 1e-5f;

// ---------------------------------------------------------------------------
// Kernel 1: per (b,c) block compute keys/queries MLPs (fp32), attention
// softmax (over n), and hq/hk projections into workspace.
// ---------------------------------------------------------------------------
DEVI void gemm256(const float* __restrict__ A, int K,
                  const float* __restrict__ W,      // global, K x 128 row-major
                  const float* __restrict__ bias,   // may be null
                  float* __restrict__ Out,          // 64 x 128 (LDS or global)
                  int act, float slope, int tid)
{
  for (int idx = tid; idx < Nn * Ss; idx += 256) {
    const int r = idx >> 7;
    const int c = idx & 127;
    float acc = bias ? bias[c] : 0.0f;
    const float* a = A + r * K;
    const float* w = W + c;
    #pragma unroll 4
    for (int k = 0; k < K; ++k) acc = fmaf(a[k], w[k * Ss], acc);
    if (act == 1)      acc = (acc >= 0.0f) ? acc : slope * acc;
    else if (act == 2) acc = tanhf(acc);
    Out[idx] = acc;
  }
}

__global__ __launch_bounds__(256) void kq_prep_kernel(
    const float* __restrict__ tL,  const float* __restrict__ tH,
    const float* __restrict__ Wk1, const float* __restrict__ bk1,
    const float* __restrict__ Wk2, const float* __restrict__ bk2,
    const float* __restrict__ Wk3, const float* __restrict__ bk3,
    const float* __restrict__ Wq1, const float* __restrict__ bq1,
    const float* __restrict__ Wq2, const float* __restrict__ bq2,
    const float* __restrict__ Wv1, const float* __restrict__ aK,
    const float* __restrict__ gK,  const float* __restrict__ betaK,
    float* __restrict__ attn, float* __restrict__ hq, float* __restrict__ hk)
{
  __shared__ __align__(16) float sX [Nn * IRc];  // 24 KB (inputs / logits)
  __shared__ __align__(16) float sT0[Nn * Ss];   // 32 KB (temp -> KEYS)
  __shared__ __align__(16) float sT1[Nn * Ss];   // 32 KB (QUERIES)
  __shared__ __align__(16) float sT2[Nn * Ss];   // 32 KB (temp)

  const int tid = threadIdx.x;
  const int bc  = blockIdx.x;
  const float slopeK = aK[0];

  // ---- queries path: tanh(tH@Wq1+bq1) @ Wq2 + bq2 -> sT1 ----
  for (int i = tid; i < Nn * IRc; i += 256) sX[i] = tH[(size_t)bc * Nn * IRc + i];
  __syncthreads();
  gemm256(sX, IRc, Wq1, bq1, sT0, /*tanh*/2, 0.0f, tid);
  __syncthreads();
  gemm256(sT0, Ss, Wq2, bq2, sT1, 0, 0.0f, tid);
  __syncthreads();

  // ---- keys path ----
  for (int i = tid; i < Nn * IRc; i += 256) sX[i] = tL[(size_t)bc * Nn * IRc + i];
  __syncthreads();
  gemm256(sX, IRc, Wk1, bk1, sT0, /*prelu*/1, slopeK, tid);
  __syncthreads();
  gemm256(sT0, Ss, Wk2, bk2, sT2, /*tanh*/2, 0.0f, tid);
  __syncthreads();
  // LayerNorm rows of sT2
  if (tid < Nn) {
    float* row = sT2 + tid * Ss;
    float s = 0.0f;
    for (int c = 0; c < Ss; ++c) s += row[c];
    const float m = s * (1.0f / Ss);
    float v = 0.0f;
    for (int c = 0; c < Ss; ++c) { float d = row[c] - m; v += d * d; }
    const float rstd = rsqrtf(v * (1.0f / Ss) + EPSf);
    for (int c = 0; c < Ss; ++c) row[c] = (row[c] - m) * rstd * gK[c] + betaK[c];
  }
  __syncthreads();
  gemm256(sT2, Ss, Wk3, bk3, sT0, 0, 0.0f, tid);   // KEYS -> sT0
  __syncthreads();

  // ---- attention logits: sL[n*64+m] = dot(keys[n], queries[m]) ----
  float* sL = sX;  // 64*64 floats (16 KB) fits in the 24 KB region
  for (int idx = tid; idx < Nn * Nn; idx += 256) {
    const int n = idx >> 6, m = idx & 63;
    const float* kr = sT0 + n * Ss;
    const float* qr = sT1 + m * Ss;
    float acc = 0.0f;
    #pragma unroll 4
    for (int s = 0; s < Ss; ++s) acc = fmaf(kr[s], qr[s], acc);
    sL[idx] = acc;
  }
  __syncthreads();
  // softmax over n (axis=2) per column m; write attn to workspace
  if (tid < Nn) {
    const int m = tid;
    float mx = -3.0e38f;
    for (int n = 0; n < Nn; ++n) mx = fmaxf(mx, sL[n * 64 + m]);
    float s = 0.0f;
    for (int n = 0; n < Nn; ++n) {
      const float e = __expf(sL[n * 64 + m] - mx);
      sL[n * 64 + m] = e;
      s += e;
    }
    const float inv = 1.0f / s;
    for (int n = 0; n < Nn; ++n)
      attn[((size_t)(bc * Nn + n)) * Nn + m] = sL[n * 64 + m] * inv;
  }
  __syncthreads();

  // ---- hq = queries @ Wv1[:S], hk = keys @ Wv1[S:] (no bias here) ----
  gemm256(sT1, Ss, Wv1,            nullptr, hq + (size_t)bc * Nn * Ss, 0, 0.0f, tid);
  gemm256(sT0, Ss, Wv1 + Ss * Ss,  nullptr, hk + (size_t)bc * Nn * Ss, 0, 0.0f, tid);
}

// ---------------------------------------------------------------------------
// Kernel 2: fused pairwise value-net with WMMA + attention-weighted reduction
// ---------------------------------------------------------------------------
// A fragment (16-bit A 16x32): lane<16 -> row=lane, K halves {k0..k0+7, k0+16..k0+23}
//                              lane>=16 -> row=lane-16, K halves {k0+8..15, k0+24..31}
DEVI v16h load_a_frag(const _Float16* stage, int lane, int k0)
{
  const int row  = lane & 15;
  const int koff = k0 + ((lane < 16) ? 0 : 8);
  const _Float16* p = stage + row * Ss + koff;
  union { v16h v; uint4 q[2]; } u;
  u.q[0] = *(const uint4*)(p);
  u.q[1] = *(const uint4*)(p + 16);
  return u.v;
}

// B fragment (16-bit B 32x16): lane<16 -> col=lane, K halves {k0..k0+7, k0+16..k0+23}
//                              lane>=16 -> col=lane-16, K halves {k0+8..15, k0+24..31}
// WT is the transposed weight: WT[col*128 + k]
DEVI v16h load_b_frag(const _Float16* WT, int lane, int j, int k0)
{
  const int col  = (j << 4) + (lane & 15);
  const int koff = k0 + ((lane < 16) ? 0 : 8);
  const _Float16* p = WT + col * Ss + koff;
  union { v16h v; uint4 q[2]; } u;
  u.q[0] = *(const uint4*)(p);
  u.q[1] = *(const uint4*)(p + 16);
  return u.v;
}

__global__ __launch_bounds__(256) void value_attn_kernel(
    const float* __restrict__ attn, const float* __restrict__ hq,
    const float* __restrict__ hk,
    const float* __restrict__ Wv2, const float* __restrict__ bv2,
    const float* __restrict__ Wv3, const float* __restrict__ bv3,
    const float* __restrict__ bv1, const float* __restrict__ aV,
    const float* __restrict__ gV,  const float* __restrict__ betaV,
    float* __restrict__ out)
{
  __shared__ __align__(16) _Float16 sW2T[Ss * Ss];    // 32 KB
  __shared__ __align__(16) _Float16 sW3T[Ss * Ss];    // 32 KB
  __shared__ __align__(16) _Float16 sA[8][16 * Ss];   // 32 KB (4 KB / wave)
  __shared__ __align__(16) float sBv1[Ss];

  const int tid  = threadIdx.x;
  const int lane = tid & 31;
  const int wave = tid >> 5;
  const int bc   = blockIdx.x >> 3;
  const int n    = ((blockIdx.x & 7) << 3) + wave;   // one n per wave

  // stage transposed f16 weights once
  for (int i = tid; i < Ss * Ss; i += 256) {
    const int k = i >> 7, c = i & 127;
    sW2T[c * Ss + k] = (_Float16)Wv2[i];
    sW3T[c * Ss + k] = (_Float16)Wv3[i];
  }
  for (int i = tid; i < Ss; i += 256) sBv1[i] = bv1[i];
  __syncthreads();

  const float av   = aV[0];
  const int   half = lane >> 4;
  const int   lm   = lane & 15;

  float gcol[8], btcol[8], b2col[8], b3col[8];
  #pragma unroll
  for (int j = 0; j < 8; ++j) {
    const int c = (j << 4) + lm;
    gcol[j] = gV[c]; btcol[j] = betaV[c]; b2col[j] = bv2[c]; b3col[j] = bv3[c];
  }

  _Float16* stage = sA[wave];
  float outacc[8];
  #pragma unroll
  for (int j = 0; j < 8; ++j) outacc[j] = 0.0f;

  const float* hkrow = hk + (size_t)(bc * Nn + n) * Ss;

  for (int m0 = 0; m0 < Nn; m0 += 16) {
    // ---- build f16 A-tile: prelu(hq[m0+row] + hk[n] + bv1) ----
    {
      const int r = lm;                     // row 0..15
      const int c0 = half * 64;             // this lane handles 64 cols
      const float* hqrow = hq + (size_t)(bc * Nn + (m0 + r)) * Ss;
      #pragma unroll
      for (int c = 0; c < 64; c += 4) {
        const float4 q4 = *(const float4*)(hqrow + c0 + c);
        const float4 k4 = *(const float4*)(hkrow + c0 + c);
        const float4 b4 = *(const float4*)(sBv1 + c0 + c);
        float x0 = q4.x + k4.x + b4.x, x1 = q4.y + k4.y + b4.y;
        float x2 = q4.z + k4.z + b4.z, x3 = q4.w + k4.w + b4.w;
        x0 = (x0 >= 0.f) ? x0 : av * x0;  x1 = (x1 >= 0.f) ? x1 : av * x1;
        x2 = (x2 >= 0.f) ? x2 : av * x2;  x3 = (x3 >= 0.f) ? x3 : av * x3;
        union { _Float16 h[4]; uint2 u; } pk;
        pk.h[0] = (_Float16)x0; pk.h[1] = (_Float16)x1;
        pk.h[2] = (_Float16)x2; pk.h[3] = (_Float16)x3;
        *(uint2*)(stage + r * Ss + c0 + c) = pk.u;
      }
    }
    asm volatile("s_wait_dscnt 0" ::: "memory");   // cross-lane LDS fence (wave-local)

    // ---- GEMM1: acc = A(16x128) @ Wv2(128x128) ----
    v8f acc[8];
    #pragma unroll
    for (int j = 0; j < 8; ++j)
      acc[j] = (v8f){0.f, 0.f, 0.f, 0.f, 0.f, 0.f, 0.f, 0.f};
    #pragma unroll
    for (int ks = 0; ks < 4; ++ks) {
      const v16h af = load_a_frag(stage, lane, ks * 32);
      #pragma unroll
      for (int j = 0; j < 8; ++j) {
        const v16h bf = load_b_frag(sW2T, lane, j, ks * 32);
        acc[j] = __builtin_amdgcn_wmma_f32_16x16x32_f16(
            false, af, false, bf, (short)0, acc[j], false, false);
      }
    }

    // ---- bias + tanh ----
    #pragma unroll
    for (int j = 0; j < 8; ++j)
      #pragma unroll
      for (int e = 0; e < 8; ++e)
        acc[j][e] = tanhf(acc[j][e] + b2col[j]);

    // ---- LayerNorm per row (row = e + 8*half); reduce over j and 16-lane group ----
    float s1[8], s2[8];
    #pragma unroll
    for (int e = 0; e < 8; ++e) { s1[e] = 0.f; s2[e] = 0.f; }
    #pragma unroll
    for (int j = 0; j < 8; ++j)
      #pragma unroll
      for (int e = 0; e < 8; ++e) { const float v = acc[j][e]; s1[e] += v; s2[e] += v * v; }
    #pragma unroll
    for (int e = 0; e < 8; ++e) {
      #pragma unroll
      for (int mask = 1; mask < 16; mask <<= 1) {
        s1[e] += __shfl_xor(s1[e], mask, 32);
        s2[e] += __shfl_xor(s2[e], mask, 32);
      }
    }
    float mean[8], rstd[8];
    #pragma unroll
    for (int e = 0; e < 8; ++e) {
      mean[e] = s1[e] * (1.0f / Ss);
      const float var = s2[e] * (1.0f / Ss) - mean[e] * mean[e];
      rstd[e] = rsqrtf(var + EPSf);
    }

    // ---- restage normalized tile as f16 A for GEMM2 ----
    #pragma unroll
    for (int j = 0; j < 8; ++j)
      #pragma unroll
      for (int e = 0; e < 8; ++e) {
        const float x = (acc[j][e] - mean[e]) * rstd[e] * gcol[j] + btcol[j];
        stage[(e + half * 8) * Ss + (j << 4) + lm] = (_Float16)x;
      }
    asm volatile("s_wait_dscnt 0" ::: "memory");

    // ---- GEMM2: values = LN(h) @ Wv3 (+bv3 folded into reduction) ----
    v8f acc2[8];
    #pragma unroll
    for (int j = 0; j < 8; ++j)
      acc2[j] = (v8f){0.f, 0.f, 0.f, 0.f, 0.f, 0.f, 0.f, 0.f};
    #pragma unroll
    for (int ks = 0; ks < 4; ++ks) {
      const v16h af = load_a_frag(stage, lane, ks * 32);
      #pragma unroll
      for (int j = 0; j < 8; ++j) {
        const v16h bf = load_b_frag(sW3T, lane, j, ks * 32);
        acc2[j] = __builtin_amdgcn_wmma_f32_16x16x32_f16(
            false, af, false, bf, (short)0, acc2[j], false, false);
      }
    }

    // ---- attn-weighted reduction over the 16 m's of this chunk ----
    float w8[8];
    const float* arow = attn + (size_t)(bc * Nn + n) * Nn + m0 + half * 8;
    #pragma unroll
    for (int e = 0; e < 8; ++e) w8[e] = arow[e];
    #pragma unroll
    for (int j = 0; j < 8; ++j) {
      float t = 0.0f;
      #pragma unroll
      for (int e = 0; e < 8; ++e) t = fmaf(w8[e], acc2[j][e] + b3col[j], t);
      outacc[j] += t;
    }
  }

  // combine the two 16-lane halves (same col, disjoint rows) and store
  #pragma unroll
  for (int j = 0; j < 8; ++j) {
    const float v = outacc[j] + __shfl_xor(outacc[j], 16, 32);
    if (half == 0)
      out[(size_t)(bc * Nn + n) * Ss + (j << 4) + lm] = v;
  }
}

// ---------------------------------------------------------------------------
extern "C" void kernel_launch(void* const* d_in, const int* in_sizes, int n_in,
                              void* d_out, int out_size, void* d_ws, size_t ws_size,
                              hipStream_t stream)
{
  const float* tL    = (const float*)d_in[0];
  const float* tH    = (const float*)d_in[1];
  const float* Wk1   = (const float*)d_in[2];
  const float* bk1   = (const float*)d_in[3];
  const float* Wk2   = (const float*)d_in[4];
  const float* bk2   = (const float*)d_in[5];
  const float* Wk3   = (const float*)d_in[6];
  const float* bk3   = (const float*)d_in[7];
  const float* Wq1   = (const float*)d_in[8];
  const float* bq1   = (const float*)d_in[9];
  const float* Wq2   = (const float*)d_in[10];
  const float* bq2   = (const float*)d_in[11];
  const float* Wv1   = (const float*)d_in[12];
  const float* bv1   = (const float*)d_in[13];
  const float* Wv2   = (const float*)d_in[14];
  const float* bv2   = (const float*)d_in[15];
  const float* Wv3   = (const float*)d_in[16];
  const float* bv3   = (const float*)d_in[17];
  const float* aK    = (const float*)d_in[18];
  const float* aV    = (const float*)d_in[19];
  const float* gK    = (const float*)d_in[20];
  const float* betaK = (const float*)d_in[21];
  const float* gV    = (const float*)d_in[22];
  const float* betaV = (const float*)d_in[23];
  float* out = (float*)d_out;

  // workspace layout (fp32): attn[BC][N][N] | hq[BC][N][S] | hk[BC][N][S]
  float* ws_attn = (float*)d_ws;
  float* ws_hq   = ws_attn + (size_t)BC * Nn * Nn;
  float* ws_hk   = ws_hq   + (size_t)BC * Nn * Ss;

  kq_prep_kernel<<<BC, 256, 0, stream>>>(
      tL, tH, Wk1, bk1, Wk2, bk2, Wk3, bk3, Wq1, bq1, Wq2, bq2,
      Wv1, aK, gK, betaK, ws_attn, ws_hq, ws_hk);

  value_attn_kernel<<<BC * 8, 256, 0, stream>>>(
      ws_attn, ws_hq, ws_hk, Wv2, bv2, Wv3, bv3, bv1, aV, gV, betaV, out);
}
